// _ENGNNBlock_60069412602312
// MI455X (gfx1250) — compile-verified, matrix operand order checked by reading, not run on hardware
//
#include <hip/hip_runtime.h>
#include <hip/hip_bf16.h>

typedef __attribute__((ext_vector_type(16))) _Float16 v16h;
typedef __attribute__((ext_vector_type(8)))  float    v8f;

#define NDIM  128
#define EDIM  128
#define HID   64
#define CAT_E 384
#define CAT_N 256

__device__ __forceinline__ v8f vzero8() {
  v8f z;
#pragma unroll
  for (int i = 0; i < 8; ++i) z[i] = 0.0f;
  return z;
}

// Load a 16x32 (f16) A/B fragment for v_wmma_f32_16x16x32_f16.
// Layout (ISA 7.12.2, 16-bit A 16x32): lane j = row/col index (lane&15),
// half = lane>>4 ; VGPR v holds K = (v<4?0:16) + half*8 + (v&3)*2 (+1).
// `base` is [16 rows][ldk] f16, row-major (works for LDS A and global f16
// pre-transposed weights B).
__device__ __forceinline__ v16h load_frag(const _Float16* base, int ldk,
                                          int kbase, int lane) {
  const int j    = lane & 15;
  const int half = lane >> 4;
  const _Float16* p = base + (size_t)j * ldk + kbase + half * 8;
  v16h f;
#pragma unroll
  for (int v = 0; v < 8; ++v) {
    const int k = ((v & 4) << 2) + ((v & 3) << 1);  // 0,2,4,6,16,18,20,22
    f[2 * v]     = p[k];
    f[2 * v + 1] = p[k + 1];
  }
  return f;
}

__device__ __forceinline__ v8f wmma16(v16h a, v16h b, v8f c) {
  // (neg_a, A, neg_b, B, c_mod, C, reuse_a, reuse_b)
  return __builtin_amdgcn_wmma_f32_16x16x32_f16(false, a, false, b, (short)0,
                                                c, false, false);
}

// ---------------------------------------------------------------------------
// Prep: zero scatter buffers; transpose+convert weights to f16 [N][K].
// ---------------------------------------------------------------------------
__global__ void prep_kernel(const float* __restrict__ ew1,
                            const float* __restrict__ ew2,
                            const float* __restrict__ nw1,
                            const float* __restrict__ nw2,
                            float* __restrict__ agg, float* __restrict__ cnt,
                            _Float16* __restrict__ w1t, _Float16* __restrict__ w2t,
                            _Float16* __restrict__ m1t, _Float16* __restrict__ m2t,
                            int n_nodes) {
  const int stride = gridDim.x * blockDim.x;
  const int i0 = blockIdx.x * blockDim.x + threadIdx.x;
  const int aggN = n_nodes * NDIM;
  for (int i = i0; i < aggN; i += stride) agg[i] = 0.0f;
  for (int i = i0; i < n_nodes; i += stride) cnt[i] = 0.0f;
  for (int i = i0; i < CAT_E * HID; i += stride)
    w1t[(i % HID) * CAT_E + (i / HID)] = (_Float16)ew1[i];
  for (int i = i0; i < HID * EDIM; i += stride)
    w2t[(i % EDIM) * HID + (i / EDIM)] = (_Float16)ew2[i];
  for (int i = i0; i < CAT_N * HID; i += stride)
    m1t[(i % HID) * CAT_N + (i / HID)] = (_Float16)nw1[i];
  for (int i = i0; i < HID * NDIM; i += stride)
    m2t[(i % NDIM) * HID + (i / NDIM)] = (_Float16)nw2[i];
}

// ---------------------------------------------------------------------------
// Edge kernel: 2 waves / block, one 16-edge tile per wave.
// e2 = e + relu([x_r|x_c|e] @ W1 + b1) @ W2 + b2 ; scatter-add e2, counts;
// LayerNorm(e2) -> out_e.
// ---------------------------------------------------------------------------
__global__ __launch_bounds__(64) void edge_kernel(
    const float* __restrict__ x, const int* __restrict__ ei,
    const float* __restrict__ eattr,
    const float* __restrict__ eb1, const float* __restrict__ eb2,
    const float* __restrict__ ne_g, const float* __restrict__ ne_b,
    const _Float16* __restrict__ w1t, const _Float16* __restrict__ w2t,
    float* __restrict__ agg, float* __restrict__ cnt,
    float* __restrict__ out_e, int NE) {
  __shared__ _Float16 s_cat[2][16][CAT_E];
  __shared__ float    s_eat[2][16][EDIM];
  __shared__ _Float16 s_h[2][16][HID];
  __shared__ int      s_col[2][16];

  const int wv   = threadIdx.x >> 5;
  const int lane = threadIdx.x & 31;
  const int e0   = (blockIdx.x * 2 + wv) * 16;

  // ---- stage cat tile (f16) + fp32 residual copy of edge_attr ----
  for (int r = 0; r < 16; ++r) {
    int e = e0 + r;
    if (e >= NE) e = NE - 1;  // clamp (stores are guarded later)
    const int ri = ei[e];
    const int ci = ei[NE + e];
    if (lane == 0) s_col[wv][r] = ci;
    const float* xr = x + (size_t)ri * NDIM;
    const float* xc = x + (size_t)ci * NDIM;
    const float* ea = eattr + (size_t)e * EDIM;
#pragma unroll
    for (int i = lane; i < NDIM; i += 32) {
      s_cat[wv][r][i]            = (_Float16)xr[i];
      s_cat[wv][r][NDIM + i]     = (_Float16)xc[i];
      const float v = ea[i];
      s_cat[wv][r][2 * NDIM + i] = (_Float16)v;
      s_eat[wv][r][i]            = v;
    }
  }
  __syncthreads();

  // ---- GEMM1: [16,384] x [384,64] ----
  v8f acc1[4];
#pragma unroll
  for (int t = 0; t < 4; ++t) acc1[t] = vzero8();
  const _Float16* catp = &s_cat[wv][0][0];
#pragma unroll
  for (int ks = 0; ks < CAT_E / 32; ++ks) {
    const v16h a = load_frag(catp, CAT_E, ks * 32, lane);
#pragma unroll
    for (int t = 0; t < 4; ++t) {
      const v16h b = load_frag(w1t + (size_t)(t * 16) * CAT_E, CAT_E, ks * 32, lane);
      acc1[t] = wmma16(a, b, acc1[t]);
    }
  }

  // ---- bias + relu -> s_h (re-layout C/D -> A) ----
  const int nl = lane & 15, hf = lane >> 4;
#pragma unroll
  for (int t = 0; t < 4; ++t) {
    const int n = t * 16 + nl;
    const float b1 = eb1[n];
#pragma unroll
    for (int r = 0; r < 8; ++r)
      s_h[wv][hf * 8 + r][n] = (_Float16)fmaxf(acc1[t][r] + b1, 0.0f);
  }
  __syncthreads();

  // ---- GEMM2: [16,64] x [64,128] ----
  v8f acc2[8];
#pragma unroll
  for (int t = 0; t < 8; ++t) acc2[t] = vzero8();
#pragma unroll
  for (int ks = 0; ks < 2; ++ks) {
    const v16h a = load_frag(&s_h[wv][0][0], HID, ks * 32, lane);
#pragma unroll
    for (int t = 0; t < 8; ++t) {
      const v16h b = load_frag(w2t + (size_t)(t * 16) * HID, HID, ks * 32, lane);
      acc2[t] = wmma16(a, b, acc2[t]);
    }
  }

  // ---- epilogue: residual + bias (fp32) ----
  float gv[8], bv[8];
#pragma unroll
  for (int t = 0; t < 8; ++t) {
    const int n = t * 16 + nl;
    const float b2 = eb2[n];
    gv[t] = ne_g[n];
    bv[t] = ne_b[n];
#pragma unroll
    for (int r = 0; r < 8; ++r)
      acc2[t][r] += b2 + s_eat[wv][hf * 8 + r][n];
  }

  // ---- scatter-add e2 into agg[col], count edges per node ----
#pragma unroll
  for (int r = 0; r < 8; ++r) {
    const int m = hf * 8 + r;
    if (e0 + m < NE) {
      const int c = s_col[wv][m];
      float* dst = agg + (size_t)c * EDIM;
#pragma unroll
      for (int t = 0; t < 8; ++t) atomicAdd(dst + t * 16 + nl, acc2[t][r]);
      if (nl == 0) atomicAdd(cnt + c, 1.0f);
    }
  }

  // ---- LayerNorm rows (16 lanes x 8 tiles per row) + store ----
#pragma unroll
  for (int r = 0; r < 8; ++r) {
    float s = 0.0f, ss = 0.0f;
#pragma unroll
    for (int t = 0; t < 8; ++t) {
      const float v = acc2[t][r];
      s += v;
      ss += v * v;
    }
#pragma unroll
    for (int off = 1; off < 16; off <<= 1) {
      s  += __shfl_xor(s, off, 32);
      ss += __shfl_xor(ss, off, 32);
    }
    const float mu   = s * (1.0f / 128.0f);
    const float var  = ss * (1.0f / 128.0f) - mu * mu;
    const float rstd = rsqrtf(var + 1e-5f);
    const int m = hf * 8 + r;
    if (e0 + m < NE) {
      float* orow = out_e + (size_t)(e0 + m) * EDIM;
#pragma unroll
      for (int t = 0; t < 8; ++t)
        orow[t * 16 + nl] = (acc2[t][r] - mu) * rstd * gv[t] + bv[t];
    }
  }
}

// ---------------------------------------------------------------------------
// Node kernel: x2 = x + relu([x | agg/max(cnt,1)] @ M1 + c1) @ M2 + c2 ;
// LayerNorm(x2) -> out_x.
// ---------------------------------------------------------------------------
__global__ __launch_bounds__(64) void node_kernel(
    const float* __restrict__ x,
    const float* __restrict__ nb1, const float* __restrict__ nb2,
    const float* __restrict__ nx_g, const float* __restrict__ nx_b,
    const _Float16* __restrict__ m1t, const _Float16* __restrict__ m2t,
    const float* __restrict__ agg, const float* __restrict__ cnt,
    float* __restrict__ out_x, int NN) {
  __shared__ _Float16 s_cat[2][16][CAT_N];
  __shared__ float    s_xr[2][16][NDIM];
  __shared__ _Float16 s_h[2][16][HID];

  const int wv   = threadIdx.x >> 5;
  const int lane = threadIdx.x & 31;
  const int n0   = (blockIdx.x * 2 + wv) * 16;

  for (int r = 0; r < 16; ++r) {
    int node = n0 + r;
    if (node >= NN) node = NN - 1;
    const float inv = 1.0f / fmaxf(cnt[node], 1.0f);
    const float* xr = x + (size_t)node * NDIM;
    const float* ar = agg + (size_t)node * NDIM;
#pragma unroll
    for (int i = lane; i < NDIM; i += 32) {
      const float xv = xr[i];
      s_cat[wv][r][i]        = (_Float16)xv;
      s_xr[wv][r][i]         = xv;
      s_cat[wv][r][NDIM + i] = (_Float16)(ar[i] * inv);
    }
  }
  __syncthreads();

  v8f acc1[4];
#pragma unroll
  for (int t = 0; t < 4; ++t) acc1[t] = vzero8();
#pragma unroll
  for (int ks = 0; ks < CAT_N / 32; ++ks) {
    const v16h a = load_frag(&s_cat[wv][0][0], CAT_N, ks * 32, lane);
#pragma unroll
    for (int t = 0; t < 4; ++t) {
      const v16h b = load_frag(m1t + (size_t)(t * 16) * CAT_N, CAT_N, ks * 32, lane);
      acc1[t] = wmma16(a, b, acc1[t]);
    }
  }

  const int nl = lane & 15, hf = lane >> 4;
#pragma unroll
  for (int t = 0; t < 4; ++t) {
    const int n = t * 16 + nl;
    const float b1 = nb1[n];
#pragma unroll
    for (int r = 0; r < 8; ++r)
      s_h[wv][hf * 8 + r][n] = (_Float16)fmaxf(acc1[t][r] + b1, 0.0f);
  }
  __syncthreads();

  v8f acc2[8];
#pragma unroll
  for (int t = 0; t < 8; ++t) acc2[t] = vzero8();
#pragma unroll
  for (int ks = 0; ks < 2; ++ks) {
    const v16h a = load_frag(&s_h[wv][0][0], HID, ks * 32, lane);
#pragma unroll
    for (int t = 0; t < 8; ++t) {
      const v16h b = load_frag(m2t + (size_t)(t * 16) * HID, HID, ks * 32, lane);
      acc2[t] = wmma16(a, b, acc2[t]);
    }
  }

  float gv[8], bv[8];
#pragma unroll
  for (int t = 0; t < 8; ++t) {
    const int n = t * 16 + nl;
    const float b2 = nb2[n];
    gv[t] = nx_g[n];
    bv[t] = nx_b[n];
#pragma unroll
    for (int r = 0; r < 8; ++r)
      acc2[t][r] += b2 + s_xr[wv][hf * 8 + r][n];
  }

#pragma unroll
  for (int r = 0; r < 8; ++r) {
    float s = 0.0f, ss = 0.0f;
#pragma unroll
    for (int t = 0; t < 8; ++t) {
      const float v = acc2[t][r];
      s += v;
      ss += v * v;
    }
#pragma unroll
    for (int off = 1; off < 16; off <<= 1) {
      s  += __shfl_xor(s, off, 32);
      ss += __shfl_xor(ss, off, 32);
    }
    const float mu   = s * (1.0f / 128.0f);
    const float var  = ss * (1.0f / 128.0f) - mu * mu;
    const float rstd = rsqrtf(var + 1e-5f);
    const int m = hf * 8 + r;
    if (n0 + m < NN) {
      float* orow = out_x + (size_t)(n0 + m) * NDIM;
#pragma unroll
      for (int t = 0; t < 8; ++t)
        orow[t * 16 + nl] = (acc2[t][r] - mu) * rstd * gv[t] + bv[t];
    }
  }
}

// ---------------------------------------------------------------------------
extern "C" void kernel_launch(void* const* d_in, const int* in_sizes, int n_in,
                              void* d_out, int out_size, void* d_ws, size_t ws_size,
                              hipStream_t stream) {
  const float* x     = (const float*)d_in[0];
  const int*   ei    = (const int*)d_in[1];
  const float* eattr = (const float*)d_in[2];
  // d_in[3] = batch (unused: single graph)
  const float* ew1 = (const float*)d_in[4];
  const float* eb1 = (const float*)d_in[5];
  const float* ew2 = (const float*)d_in[6];
  const float* eb2 = (const float*)d_in[7];
  const float* nw1 = (const float*)d_in[8];
  const float* nb1 = (const float*)d_in[9];
  const float* nw2 = (const float*)d_in[10];
  const float* nb2 = (const float*)d_in[11];
  const float* nx_g = (const float*)d_in[12];
  const float* nx_b = (const float*)d_in[13];
  const float* ne_g = (const float*)d_in[14];
  const float* ne_b = (const float*)d_in[15];

  const int NN = in_sizes[0] / NDIM;
  const int NE = in_sizes[2] / EDIM;

  // workspace layout
  char* ws = (char*)d_ws;
  size_t off = 0;
  float* agg = (float*)(ws + off);  off += (size_t)NN * NDIM * sizeof(float);
  off = (off + 255) & ~(size_t)255;
  float* cnt = (float*)(ws + off);  off += (size_t)NN * sizeof(float);
  off = (off + 255) & ~(size_t)255;
  _Float16* w1t = (_Float16*)(ws + off); off += (size_t)CAT_E * HID * 2;
  off = (off + 255) & ~(size_t)255;
  _Float16* w2t = (_Float16*)(ws + off); off += (size_t)HID * EDIM * 2;
  off = (off + 255) & ~(size_t)255;
  _Float16* m1t = (_Float16*)(ws + off); off += (size_t)CAT_N * HID * 2;
  off = (off + 255) & ~(size_t)255;
  _Float16* m2t = (_Float16*)(ws + off); off += (size_t)HID * NDIM * 2;

  float* out_x = (float*)d_out;
  float* out_e = out_x + (size_t)NN * NDIM;

  prep_kernel<<<2048, 256, 0, stream>>>(ew1, ew2, nw1, nw2, agg, cnt, w1t, w2t,
                                        m1t, m2t, NN);
  const int egrid = (NE + 31) / 32;  // 32 edges per block (2 waves x 16)
  edge_kernel<<<egrid, 64, 0, stream>>>(x, ei, eattr, eb1, eb2, ne_g, ne_b, w1t,
                                        w2t, agg, cnt, out_e, NE);
  const int ngrid = (NN + 31) / 32;  // 32 nodes per block
  node_kernel<<<ngrid, 64, 0, stream>>>(x, nb1, nb2, nx_g, nx_b, m1t, m2t, agg,
                                        cnt, out_x, NN);
}
